// InContextBlock_14302241095728
// MI455X (gfx1250) — compile-verified, matrix operand order checked by reading, not run on hardware
//
#include <hip/hip_runtime.h>
#include <hip/hip_bf16.h>

// ---------------------------------------------------------------------------
// Problem constants (match the reference)
// ---------------------------------------------------------------------------
namespace cfg {
constexpr int B   = 2;
constexpr int N   = 2048;
constexpr int C   = 1024;
constexpr int H   = 16;
constexpr int HD  = 64;          // C / H
constexpr int HID = 2730;        // int(C*4*2/3)
constexpr int HIDP = 2752;       // HID padded to multiple of 64 (for WMMA K loop)
constexpr int M   = B * N;       // 4096 token rows
constexpr float EPS = 1e-6f;
}

typedef __attribute__((ext_vector_type(16))) __bf16 v16bf;
typedef __attribute__((ext_vector_type(8)))  __bf16 v8bf;
typedef __attribute__((ext_vector_type(8)))  float  v8f;

__device__ __forceinline__ v16bf cat8(v8bf lo, v8bf hi) {
  return __builtin_shufflevector(lo, hi, 0,1,2,3,4,5,6,7,8,9,10,11,12,13,14,15);
}
__device__ __forceinline__ v8bf load8(const __bf16* p) {
  return *(const v8bf*)p;
}
__device__ __forceinline__ v8f zero8f() {
  v8f z;
#pragma unroll
  for (int i = 0; i < 8; ++i) z[i] = 0.f;
  return z;
}
__device__ __forceinline__ v8f wmma_bf16(v16bf a, v16bf b, v8f c) {
  return __builtin_amdgcn_wmma_f32_16x16x32_bf16(
      /*neg_a=*/false, a, /*neg_b=*/false, b,
      /*c_mod=*/(short)0, c, /*reuse_a=*/false, /*reuse_b=*/false);
}

// ---------------------------------------------------------------------------
// f32 -> bf16 staging copy with column padding (pad cols write 0)
// ---------------------------------------------------------------------------
__global__ __launch_bounds__(256)
void cvt_bf16_kernel(const float* __restrict__ src, __bf16* __restrict__ dst,
                     int rows, int cols, int ldd) {
  int total = rows * ldd;
  for (int idx = blockIdx.x * 256 + threadIdx.x; idx < total;
       idx += gridDim.x * 256) {
    int r = idx / ldd;
    int c = idx - r * ldd;
    float v = (c < cols) ? src[(size_t)r * cols + c] : 0.f;
    dst[idx] = (__bf16)v;
  }
}

// ---------------------------------------------------------------------------
// RMSNorm over last dim (C), one block per row. Writes f32 and/or bf16.
// ---------------------------------------------------------------------------
__global__ __launch_bounds__(256)
void rmsnorm_kernel(const float* __restrict__ x, const float* __restrict__ w,
                    float* __restrict__ outf, __bf16* __restrict__ outb,
                    int Cc, float eps) {
  const int row = blockIdx.x;
  const float* xr = x + (size_t)row * Cc;
  float ss = 0.f;
  for (int c = threadIdx.x; c < Cc; c += 256) {
    float v = xr[c];
    ss += v * v;
  }
#pragma unroll
  for (int o = 16; o > 0; o >>= 1) ss += __shfl_xor(ss, o, 32);
  __shared__ float red[8];
  if ((threadIdx.x & 31) == 0) red[threadIdx.x >> 5] = ss;
  __syncthreads();
  float tot = 0.f;
#pragma unroll
  for (int i = 0; i < 8; ++i) tot += red[i];
  float sc = rsqrtf(tot / (float)Cc + eps);
  for (int c = threadIdx.x; c < Cc; c += 256) {
    float v = xr[c] * sc * w[c];
    if (outf) outf[(size_t)row * Cc + c] = v;
    if (outb) outb[(size_t)row * Cc + c] = (__bf16)v;
  }
}

// ---------------------------------------------------------------------------
// Generic bf16 WMMA GEMM:  out[M,N] = A[M,K] * W[N,K]^T (+bias) (+residual)
// Block: 256 threads = 8 waves (4 row-waves x 2 col-waves), tile 128x128.
// Wave tile: 32(M) x 64(N) = 2x4 grid of 16x16 WMMA accumulators.
// N-edge handled by CLAMPING weight row addresses (no exec masking in the
// hot loop) and masking at store time.  K must be a multiple of 32 (caller
// pads).  M must be a multiple of 128.
// ---------------------------------------------------------------------------
__global__ __launch_bounds__(256)
void gemm_bf16_kernel(const __bf16* __restrict__ A, int lda,
                      const __bf16* __restrict__ W, int ldw,
                      const float* __restrict__ bias,
                      const float* __restrict__ res,
                      float* __restrict__ out, int ldo,
                      int Nn, int Kk) {
  const int lane = threadIdx.x & 31;
  const int wave = threadIdx.x >> 5;
  const int g    = lane >> 4;     // K-group half
  const int lm   = lane & 15;     // row-within-tile (A) / col-within-tile (B)

  const int rowBase = blockIdx.y * 128 + (wave & 3) * 32;
  const int colBase = blockIdx.x * 128 + (wave >> 2) * 64;

  v8f acc[2][4];
#pragma unroll
  for (int i = 0; i < 2; ++i)
#pragma unroll
    for (int j = 0; j < 4; ++j) acc[i][j] = zero8f();

  const __bf16* ap[2];
  ap[0] = A + (size_t)(rowBase + lm) * lda + g * 8;
  ap[1] = ap[0] + (size_t)16 * lda;

  const __bf16* wp[4];
  int cidx[4];
#pragma unroll
  for (int j = 0; j < 4; ++j) {
    int c = colBase + j * 16 + lm;
    cidx[j] = c;
    int cc = c < Nn ? c : (Nn - 1);    // clamp: always-valid address
    wp[j] = W + (size_t)cc * ldw + g * 8;
  }

#pragma unroll 1
  for (int k0 = 0; k0 < Kk; k0 += 32) {
    if (k0 + 128 < Kk) {               // uniform scalar guard
      __builtin_prefetch(ap[0] + k0 + 128, 0, 1);
      __builtin_prefetch(ap[1] + k0 + 128, 0, 1);
      __builtin_prefetch(wp[0] + k0 + 128, 0, 1);
      __builtin_prefetch(wp[1] + k0 + 128, 0, 1);
      __builtin_prefetch(wp[2] + k0 + 128, 0, 1);
      __builtin_prefetch(wp[3] + k0 + 128, 0, 1);
    }
    v16bf a[2], b[4];
#pragma unroll
    for (int i = 0; i < 2; ++i)
      a[i] = cat8(load8(ap[i] + k0), load8(ap[i] + k0 + 16));
#pragma unroll
    for (int j = 0; j < 4; ++j)
      b[j] = cat8(load8(wp[j] + k0), load8(wp[j] + k0 + 16));
#pragma unroll
    for (int i = 0; i < 2; ++i)
#pragma unroll
      for (int j = 0; j < 4; ++j)
        acc[i][j] = wmma_bf16(a[i], b[j], acc[i][j]);
  }

#pragma unroll
  for (int j = 0; j < 4; ++j) {
    int col = cidx[j];
    if (col >= Nn) continue;
    float bv = bias ? bias[col] : 0.f;
#pragma unroll
    for (int i = 0; i < 2; ++i) {
#pragma unroll
      for (int r = 0; r < 8; ++r) {
        int row = rowBase + i * 16 + r + 8 * g;
        float v = acc[i][j][r] + bv;
        if (res) v += res[(size_t)row * ldo + col];
        out[(size_t)row * ldo + col] = v;
      }
    }
  }
}

// ---------------------------------------------------------------------------
// QKV post-process: per (token, head): rmsnorm(q)*qw*scale, rmsnorm(k)*kw,
// pass v; writes q,k in [B,H,N,HD] bf16 and v transposed [B,H,HD,N] bf16.
// One wave (32 threads) per (token, head); 2 elems per lane.
// ---------------------------------------------------------------------------
__global__ __launch_bounds__(32)
void qkv_post_kernel(const float* __restrict__ qkv,
                     const float* __restrict__ qw, const float* __restrict__ kw,
                     __bf16* __restrict__ qb, __bf16* __restrict__ kb,
                     __bf16* __restrict__ vt,
                     int Nn, int Hh, float eps, float scale) {
  const int t = blockIdx.x;        // b*N + n
  const int h = blockIdx.y;
  const int lane = threadIdx.x;
  const int b = t / Nn;
  const int n = t - b * Nn;
  const int bh = b * Hh + h;
  const int d0 = lane * 2;
  const float* row = qkv + (size_t)t * (3 * cfg::C);

  float q0 = row[h * 64 + d0],              q1 = row[h * 64 + d0 + 1];
  float k0 = row[cfg::C + h * 64 + d0],     k1 = row[cfg::C + h * 64 + d0 + 1];
  float v0 = row[2 * cfg::C + h * 64 + d0], v1 = row[2 * cfg::C + h * 64 + d0 + 1];

  float sq = q0 * q0 + q1 * q1;
  float sk = k0 * k0 + k1 * k1;
#pragma unroll
  for (int o = 16; o > 0; o >>= 1) {
    sq += __shfl_xor(sq, o, 32);
    sk += __shfl_xor(sk, o, 32);
  }
  float scq = rsqrtf(sq / 64.f + eps) * scale;
  float sck = rsqrtf(sk / 64.f + eps);

  size_t qkOff = ((size_t)bh * Nn + n) * 64 + d0;
  qb[qkOff]     = (__bf16)(q0 * scq * qw[d0]);
  qb[qkOff + 1] = (__bf16)(q1 * scq * qw[d0 + 1]);
  kb[qkOff]     = (__bf16)(k0 * sck * kw[d0]);
  kb[qkOff + 1] = (__bf16)(k1 * sck * kw[d0 + 1]);
  vt[((size_t)bh * 64 + d0)     * Nn + n] = (__bf16)v0;
  vt[((size_t)bh * 64 + d0 + 1) * Nn + n] = (__bf16)v1;
}

// ---------------------------------------------------------------------------
// Flash attention: one workgroup (4 waves) per (b, h, 16-row q tile).
// Scores via WMMA into f32 LDS (key blocks of 512); online softmax writes
// probabilities as bf16 into a second LDS tile; P@V via WMMA reads bf16
// fragments straight from LDS (no per-element converts in phase C).
// q,k: [B,H,N,HD] bf16 (scale folded into q); v: [B,H,HD,N] bf16 transposed.
// Output written as bf16 in [B,N,C] layout (feeds proj GEMM A operand).
// ---------------------------------------------------------------------------
__global__ __launch_bounds__(128)
void attn_kernel(const __bf16* __restrict__ qb, const __bf16* __restrict__ kb,
                 const __bf16* __restrict__ vt, __bf16* __restrict__ ob,
                 int Nn, int Hh, int Cc) {
  constexpr int KB = 512;                             // keys per block
  __shared__ float S[16][KB];                         // 32 KB scores
  __shared__ __attribute__((aligned(16))) __bf16 P[16][KB];  // 16 KB probs
  __shared__ float m_s[16], l_s[16], f_s[16];

  const int qt   = blockIdx.x;          // q tile (16 rows)
  const int h    = blockIdx.y;
  const int b    = blockIdx.z;
  const int bh   = b * Hh + h;
  const int lane = threadIdx.x & 31;
  const int wave = threadIdx.x >> 5;
  const int g    = lane >> 4;
  const int lm   = lane & 15;

  if (threadIdx.x < 16) { m_s[threadIdx.x] = -3.0e38f; l_s[threadIdx.x] = 0.f; }
  __syncthreads();

  // Q fragments for this tile (rows qt*16..+15), K-dim = HD = 64 -> 2 steps
  const __bf16* qp = qb + ((size_t)bh * Nn + qt * 16 + lm) * 64 + g * 8;
  v16bf aq0 = cat8(load8(qp),      load8(qp + 16));
  v16bf aq1 = cat8(load8(qp + 32), load8(qp + 48));

  v8f acc = zero8f();                   // o tile: 16 q-rows x 16 d-cols (per wave)

  const int nBlocks = Nn / KB;
  for (int kbI = 0; kbI < nBlocks; ++kbI) {
    const int keyBase = kbI * KB;

    // ---- Phase A: scores into LDS ------------------------------------
#pragma unroll 1
    for (int i = 0; i < KB / 16 / 4; ++i) {
      int ntl = wave + 4 * i;           // local 16-key tile
      int key = keyBase + ntl * 16 + lm;
      const __bf16* kp = kb + ((size_t)bh * Nn + key) * 64 + g * 8;
      v16bf b0 = cat8(load8(kp),      load8(kp + 16));
      v16bf b1 = cat8(load8(kp + 32), load8(kp + 48));
      v8f s = zero8f();
      s = wmma_bf16(aq0, b0, s);
      s = wmma_bf16(aq1, b1, s);
#pragma unroll
      for (int r = 0; r < 8; ++r) S[r + 8 * g][ntl * 16 + lm] = s[r];
    }
    __syncthreads();

    // ---- Phase B: online softmax over this key block ------------------
    {
      const int row = threadIdx.x >> 3;  // 0..15
      const int sub = threadIdx.x & 7;   // 0..7 (contiguous lanes, same wave)
      float bm = -3.0e38f;
      for (int c = sub; c < KB; c += 8) bm = fmaxf(bm, S[row][c]);
#pragma unroll
      for (int o = 1; o < 8; o <<= 1) bm = fmaxf(bm, __shfl_xor(bm, o, 8));
      float mold = m_s[row];
      float mnew = fmaxf(mold, bm);
      float fsc  = __expf(mold - mnew);
      float ps = 0.f;
      for (int c = sub; c < KB; c += 8) {
        float p = __expf(S[row][c] - mnew);
        P[row][c] = (__bf16)p;
        ps += p;
      }
#pragma unroll
      for (int o = 1; o < 8; o <<= 1) ps += __shfl_xor(ps, o, 8);
      if (sub == 0) {
        m_s[row] = mnew;
        l_s[row] = l_s[row] * fsc + ps;
        f_s[row] = fsc;
      }
    }
    __syncthreads();

    // ---- Phase C: rescale and accumulate P @ V ------------------------
#pragma unroll
    for (int r = 0; r < 8; ++r) acc[r] *= f_s[r + 8 * g];

    const __bf16* vp = vt + ((size_t)bh * 64 + wave * 16 + lm) * Nn
                          + keyBase + g * 8;
#pragma unroll 1
    for (int kt = 0; kt < KB; kt += 32) {
      const __bf16* pp = &P[lm][kt + g * 8];
      v16bf pa = cat8(*(const v8bf*)pp, *(const v8bf*)(pp + 16));
      v16bf bv = cat8(load8(vp + kt), load8(vp + kt + 16));
      acc = wmma_bf16(pa, bv, acc);
    }
    __syncthreads();                    // S/P reused next block
  }

  // ---- Store: o[b, n, h*64 + d] in bf16 -------------------------------
#pragma unroll
  for (int r = 0; r < 8; ++r) {
    int qrow = qt * 16 + r + 8 * g;
    float v = acc[r] / l_s[r + 8 * g];
    ob[((size_t)(b * Nn + qrow)) * Cc + h * 64 + wave * 16 + lm] = (__bf16)v;
  }
}

// ---------------------------------------------------------------------------
// SwiGLU: hidden[m,j] = silu(x12[m,j]) * x12[m,HID+j]; bf16 out, K-padded.
// ---------------------------------------------------------------------------
__global__ __launch_bounds__(256)
void swiglu_kernel(const float* __restrict__ x12, __bf16* __restrict__ hid,
                   int Mm, int hidv, int hidp) {
  int total = Mm * hidp;
  for (int idx = blockIdx.x * 256 + threadIdx.x; idx < total;
       idx += gridDim.x * 256) {
    int m = idx / hidp;
    int j = idx - m * hidp;
    float r = 0.f;
    if (j < hidv) {
      float a  = x12[(size_t)m * (2 * hidv) + j];
      float b2 = x12[(size_t)m * (2 * hidv) + hidv + j];
      r = (a / (1.f + __expf(-a))) * b2;
    }
    hid[idx] = (__bf16)r;
  }
}

// ---------------------------------------------------------------------------
// Launch orchestration
// ---------------------------------------------------------------------------
extern "C" void kernel_launch(void* const* d_in, const int* in_sizes, int n_in,
                              void* d_out, int out_size, void* d_ws, size_t ws_size,
                              hipStream_t stream) {
  using namespace cfg;
  const float* x      = (const float*)d_in[0];
  const float* aln_w  = (const float*)d_in[1];
  const float* qkv_w  = (const float*)d_in[2];
  const float* qkv_b  = (const float*)d_in[3];
  const float* qn_w   = (const float*)d_in[4];
  const float* kn_w   = (const float*)d_in[5];
  const float* proj_w = (const float*)d_in[6];
  const float* proj_b = (const float*)d_in[7];
  const float* mln_w  = (const float*)d_in[8];
  const float* w12_w  = (const float*)d_in[9];
  const float* w12_b  = (const float*)d_in[10];
  const float* w3_w   = (const float*)d_in[11];
  const float* w3_b   = (const float*)d_in[12];
  float* out = (float*)d_out;

  // workspace carve (256B aligned)
  char* p = (char*)d_ws;
  auto carve = [&](size_t bytes) -> char* {
    char* r = p;
    p += (bytes + 255) & ~(size_t)255;
    return r;
  };
  float*  xn_f   = (float*) carve((size_t)M * C * 4);          // normed x (residual)
  __bf16* xn_b   = (__bf16*)carve((size_t)M * C * 2);
  __bf16* qkvw_b = (__bf16*)carve((size_t)3 * C * C * 2);
  float*  qkv_f  = (float*) carve((size_t)M * 3 * C * 4);
  __bf16* q_b    = (__bf16*)carve((size_t)B * H * N * HD * 2);
  __bf16* k_b    = (__bf16*)carve((size_t)B * H * N * HD * 2);
  __bf16* v_t    = (__bf16*)carve((size_t)B * H * HD * N * 2); // transposed
  __bf16* o_b    = (__bf16*)carve((size_t)M * C * 2);
  __bf16* projw_b= (__bf16*)carve((size_t)C * C * 2);
  float*  x_mid  = (float*) carve((size_t)M * C * 4);
  __bf16* h_b    = (__bf16*)carve((size_t)M * C * 2);
  __bf16* w12w_b = (__bf16*)carve((size_t)2 * HID * C * 2);
  float*  x12_f  = (float*) carve((size_t)M * 2 * HID * 4);
  __bf16* hid_b  = (__bf16*)carve((size_t)M * HIDP * 2);
  __bf16* w3w_b  = (__bf16*)carve((size_t)C * HIDP * 2);
  (void)ws_size; (void)in_sizes; (void)n_in; (void)out_size;

  // ---- weight staging to bf16 (with K padding for w3) ----
  cvt_bf16_kernel<<<2048, 256, 0, stream>>>(qkv_w, qkvw_b, 3 * C, C, C);
  cvt_bf16_kernel<<<1024, 256, 0, stream>>>(proj_w, projw_b, C, C, C);
  cvt_bf16_kernel<<<4096, 256, 0, stream>>>(w12_w, w12w_b, 2 * HID, C, C);
  cvt_bf16_kernel<<<2048, 256, 0, stream>>>(w3_w, w3w_b, C, HID, HIDP);

  // ---- attn pre-norm ----
  rmsnorm_kernel<<<M, 256, 0, stream>>>(x, aln_w, xn_f, xn_b, C, EPS);

  // ---- qkv = xn @ qkv_w^T + b ----
  gemm_bf16_kernel<<<dim3((3 * C) / 128, M / 128), 256, 0, stream>>>(
      xn_b, C, qkvw_b, C, qkv_b, nullptr, qkv_f, 3 * C, 3 * C, C);

  // ---- per-head q/k rmsnorm + scale, v transpose ----
  qkv_post_kernel<<<dim3(M, H), 32, 0, stream>>>(
      qkv_f, qn_w, kn_w, q_b, k_b, v_t, N, H, EPS, 0.125f /* 1/sqrt(64) */);

  // ---- attention ----
  attn_kernel<<<dim3(N / 16, H, B), 128, 0, stream>>>(q_b, k_b, v_t, o_b, N, H, C);

  // ---- x_mid = xn + o @ proj_w^T + b ----
  gemm_bf16_kernel<<<dim3(C / 128, M / 128), 256, 0, stream>>>(
      o_b, C, projw_b, C, proj_b, xn_f, x_mid, C, C, C);

  // ---- mlp pre-norm ----
  rmsnorm_kernel<<<M, 256, 0, stream>>>(x_mid, mln_w, nullptr, h_b, C, EPS);

  // ---- x12 = h @ w12_w^T + b  (N = 5460, clamped edge) ----
  gemm_bf16_kernel<<<dim3((2 * HID + 127) / 128, M / 128), 256, 0, stream>>>(
      h_b, C, w12w_b, C, w12_b, nullptr, x12_f, 2 * HID, 2 * HID, C);

  // ---- swiglu -> bf16 hidden (K-padded to HIDP) ----
  swiglu_kernel<<<(M * HIDP + 255) / 256, 256, 0, stream>>>(
      x12_f, hid_b, M, HID, HIDP);

  // ---- out = x_mid + hidden @ w3_w^T + b ----
  gemm_bf16_kernel<<<dim3(C / 128, M / 128), 256, 0, stream>>>(
      hid_b, HIDP, w3w_b, HIDP, w3_b, x_mid, out, C, C, HIDP);
}
